// MultiHeadAttention_88381837017264
// MI455X (gfx1250) — compile-verified, matrix operand order checked by reading, not run on hardware
//
#include <hip/hip_runtime.h>
#include <hip/hip_bf16.h>

// ---------------------------------------------------------------------------
// MultiHeadAttention forward, MI455X (gfx1250), wave32 + WMMA + async-to-LDS.
// proj(Q,K,V) -> scores(+masks) -> softmax -> PV -> out-proj(+bias)
// GEMMs: v_wmma_f32_16x16x32_f16 (f16 in, f32 accumulate).
// f16 operands stream via GLOBAL_LOAD_ASYNC_TO_LDS_B128 (ASYNCcnt);
// f32 operands are register-prefetched and converted to f16 into LDS.
// All K-loops are double-buffered: tile k+1 is in flight under tile k's WMMAs.
// ---------------------------------------------------------------------------

typedef __attribute__((ext_vector_type(16))) _Float16 v16h;
typedef __attribute__((ext_vector_type(8)))  _Float16 v8h;
typedef __attribute__((ext_vector_type(8)))  float    v8f;

#define SEQ    2048
#define BATCH  2
#define DMODEL 1024
#define NHEAD  16
#define HDIM   64
#define NEGBIG (-1.0e9f)

// 256 threads = 8 waves; block tile 128x64; wave tile 32x32 (2x2 frags); K-tile 64.
#define BM 128
#define BN 64
#define BK 64
#define TSTR 72                 // LDS row stride in halves (144B: 16B aligned)
#define ATILE (BM * TSTR)       // 9216 halves
#define BTILE (BN * TSTR)       // 4608 halves

#define WMMA_F16(A_, B_, C_) \
  __builtin_amdgcn_wmma_f32_16x16x32_f16(false, (A_), false, (B_), (short)0, (C_), false, false)

// ---- async global->LDS (CDNA5 ASYNCcnt path), GV mode ---------------------

__device__ __forceinline__ uint32_t lds_addr32(const _Float16* p) {
  // FLAT aperture rule: low 32 bits of a generic LDS address are the
  // wave-relative LDS byte address (ISA 10.2).
  return (uint32_t)(uintptr_t)p;
}

// A tile: BM(128) rows x BK(64) halves; thread -> row=tid>>1, col=(tid&1)*32.
__device__ __forceinline__ void ldA_f16_async(const _Float16* g, int ld, int tid,
                                              _Float16* sA) {
  int row = tid >> 1;
  int col = (tid & 1) << 5;
  const _Float16* src = g + (size_t)row * ld + col;
  uint32_t lds = lds_addr32(sA + row * TSTR + col);
  asm volatile(
      "global_load_async_to_lds_b128 %0, %1, off\n"
      "global_load_async_to_lds_b128 %0, %1, off offset:16\n"
      "global_load_async_to_lds_b128 %0, %1, off offset:32\n"
      "global_load_async_to_lds_b128 %0, %1, off offset:48"
      :: "v"(lds), "v"(src) : "memory");
}

// B tile: BN(64) rows x BK(64) halves; thread -> row=tid>>2, col=(tid&3)*16.
__device__ __forceinline__ void ldB_f16_async(const _Float16* g, int ld, int tid,
                                              _Float16* sB) {
  int row = tid >> 2;
  int col = (tid & 3) << 4;
  const _Float16* src = g + (size_t)row * ld + col;
  uint32_t lds = lds_addr32(sB + row * TSTR + col);
  asm volatile(
      "global_load_async_to_lds_b128 %0, %1, off\n"
      "global_load_async_to_lds_b128 %0, %1, off offset:16"
      :: "v"(lds), "v"(src) : "memory");
}

__device__ __forceinline__ void wait_async0() {
  asm volatile("s_wait_asynccnt 0x0" ::: "memory");
}

// ---- f32 register-prefetch + f16 convert/store ----------------------------

struct ARegs { float4 f[8]; };   // 32 floats per thread
struct BRegs { float4 f[4]; };   // 16 floats per thread

__device__ __forceinline__ ARegs ldA_f32_regs(const float* g, int ld, int tid) {
  int row = tid >> 1, col = (tid & 1) << 5;
  const float4* src = (const float4*)(g + (size_t)row * ld + col);
  ARegs r;
#pragma unroll
  for (int i = 0; i < 8; ++i) r.f[i] = src[i];
  return r;
}

__device__ __forceinline__ void stA_f16(const ARegs& r, int tid, _Float16* sA) {
  int row = tid >> 1, col = (tid & 1) << 5;
  const float* fp = (const float*)r.f;
  _Float16* d = sA + row * TSTR + col;
#pragma unroll
  for (int c = 0; c < 4; ++c) {
    v8h h;
#pragma unroll
    for (int i = 0; i < 8; ++i) h[i] = (_Float16)fp[c * 8 + i];
    *(v8h*)(d + c * 8) = h;
  }
}

__device__ __forceinline__ BRegs ldB_f32_regs(const float* g, int ld, int tid) {
  int row = tid >> 2, col = (tid & 3) << 4;
  const float4* src = (const float4*)(g + (size_t)row * ld + col);
  BRegs r;
#pragma unroll
  for (int i = 0; i < 4; ++i) r.f[i] = src[i];
  return r;
}

__device__ __forceinline__ void stB_f16(const BRegs& r, int tid, _Float16* sB) {
  int row = tid >> 2, col = (tid & 3) << 4;
  const float* fp = (const float*)r.f;
  _Float16* d = sB + row * TSTR + col;
#pragma unroll
  for (int c = 0; c < 2; ++c) {
    v8h h;
#pragma unroll
    for (int i = 0; i < 8; ++i) h[i] = (_Float16)fp[c * 8 + i];
    *(v8h*)(d + c * 8) = h;
  }
}

// ---- LDS -> WMMA fragment (ISA 7.12.2 16-bit 16xK layout) -----------------

__device__ __forceinline__ v16h frag64(const _Float16* t, int row0, int kk, int lane) {
  const _Float16* p = t + (row0 + (lane & 15)) * TSTR + kk + ((lane >> 4) << 3);
  v8h lo = *(const v8h*)p;
  v8h hi = *(const v8h*)(p + 16);
  v16h r;
#pragma unroll
  for (int i = 0; i < 8; ++i) { r[i] = lo[i]; r[8 + i] = hi[i]; }
  return r;
}

__device__ __forceinline__ void mma_step(const _Float16* sA, const _Float16* sB,
                                         int wm, int wn, int lane, v8f acc[2][2]) {
#pragma unroll
  for (int kk = 0; kk < BK; kk += 32) {
    v16h a0 = frag64(sA, wm,      kk, lane);
    v16h a1 = frag64(sA, wm + 16, kk, lane);
    v16h b0 = frag64(sB, wn,      kk, lane);
    v16h b1 = frag64(sB, wn + 16, kk, lane);
    acc[0][0] = WMMA_F16(a0, b0, acc[0][0]);
    acc[0][1] = WMMA_F16(a0, b1, acc[0][1]);
    acc[1][0] = WMMA_F16(a1, b0, acc[1][0]);
    acc[1][1] = WMMA_F16(a1, b1, acc[1][1]);
  }
}

// Stage f32 accumulators into LDS as f16 tile [BM][TSTR].
__device__ __forceinline__ void stage_acc(_Float16* st, int wm, int wn, int lane,
                                          const v8f acc[2][2]) {
  int n0 = lane & 15, mh = (lane >> 4) << 3;
#pragma unroll
  for (int i = 0; i < 2; ++i)
#pragma unroll
    for (int j = 0; j < 2; ++j)
#pragma unroll
      for (int r = 0; r < 8; ++r)
        st[(wm + i * 16 + mh + r) * TSTR + wn + j * 16 + n0] = (_Float16)acc[i][j][r];
}

__device__ __forceinline__ void store_row32(_Float16* dst, const _Float16* src) {
  v8h x0 = *(const v8h*)(src);
  v8h x1 = *(const v8h*)(src + 8);
  v8h x2 = *(const v8h*)(src + 16);
  v8h x3 = *(const v8h*)(src + 24);
  v8h* d = (v8h*)dst;
  d[0] = x0; d[1] = x1; d[2] = x2; d[3] = x3;
}

// ---------------------------------------------------------------------------
// Kernel 1a: Q/K projection, double-buffered f32 GEMM, head-major f16 out.
// BN==HDIM so h == blockIdx.x for the whole block.
// ---------------------------------------------------------------------------
__global__ void __launch_bounds__(256) proj_qk_kernel(const float* __restrict__ X,
                                                      const float* __restrict__ W,
                                                      _Float16* __restrict__ out) {
  __shared__ __align__(16) _Float16 smem[2 * ATILE + 2 * BTILE];
  int tid = threadIdx.x, lane = tid & 31, wave = tid >> 5;
  int wm = (wave >> 1) << 5, wn = (wave & 1) << 5;
  const float* Ab = X + (size_t)blockIdx.y * BM * DMODEL;
  const float* Bb = W + (size_t)blockIdx.x * BN * DMODEL;
  // Prologue: tile 0 into buffer 0.
  {
    ARegs ar = ldA_f32_regs(Ab, DMODEL, tid);
    BRegs br = ldB_f32_regs(Bb, DMODEL, tid);
    stA_f16(ar, tid, smem);
    stB_f16(br, tid, smem + 2 * ATILE);
  }
  __syncthreads();
  v8f acc[2][2] = {};
  int buf = 0;
  for (int k0 = 0; k0 < DMODEL; k0 += BK) {
    int nb = buf ^ 1;
    bool more = (k0 + BK) < DMODEL;
    ARegs ar; BRegs br;
    if (more) {
      ar = ldA_f32_regs(Ab + k0 + BK, DMODEL, tid);
      br = ldB_f32_regs(Bb + k0 + BK, DMODEL, tid);
    }
    mma_step(smem + buf * ATILE, smem + 2 * ATILE + buf * BTILE, wm, wn, lane, acc);
    if (more) {
      stA_f16(ar, tid, smem + nb * ATILE);
      stB_f16(br, tid, smem + 2 * ATILE + nb * BTILE);
    }
    __syncthreads();
    buf = nb;
  }
  stage_acc(smem, wm, wn, lane, acc);
  __syncthreads();
  int m = tid >> 1, half = tid & 1;
  int gm = blockIdx.y * BM + m;            // token = s*B + b
  int s = gm >> 1, b = gm & 1, h = blockIdx.x;
  store_row32(out + ((size_t)(b * NHEAD + h) * SEQ + s) * HDIM + half * 32,
              smem + m * TSTR + half * 32);
}

// ---------------------------------------------------------------------------
// Kernel 1b: V projection, written transposed per head: Vt[bh][dk][S].
// ---------------------------------------------------------------------------
__global__ void __launch_bounds__(256) proj_v_kernel(const float* __restrict__ X,
                                                     const float* __restrict__ W,
                                                     _Float16* __restrict__ out) {
  __shared__ __align__(16) _Float16 smem[2 * ATILE + 2 * BTILE];
  int tid = threadIdx.x, lane = tid & 31, wave = tid >> 5;
  int wm = (wave >> 1) << 5, wn = (wave & 1) << 5;
  const float* Ab = X + (size_t)blockIdx.y * BM * DMODEL;
  const float* Bb = W + (size_t)blockIdx.x * BN * DMODEL;
  {
    ARegs ar = ldA_f32_regs(Ab, DMODEL, tid);
    BRegs br = ldB_f32_regs(Bb, DMODEL, tid);
    stA_f16(ar, tid, smem);
    stB_f16(br, tid, smem + 2 * ATILE);
  }
  __syncthreads();
  v8f acc[2][2] = {};
  int buf = 0;
  for (int k0 = 0; k0 < DMODEL; k0 += BK) {
    int nb = buf ^ 1;
    bool more = (k0 + BK) < DMODEL;
    ARegs ar; BRegs br;
    if (more) {
      ar = ldA_f32_regs(Ab + k0 + BK, DMODEL, tid);
      br = ldB_f32_regs(Bb + k0 + BK, DMODEL, tid);
    }
    mma_step(smem + buf * ATILE, smem + 2 * ATILE + buf * BTILE, wm, wn, lane, acc);
    if (more) {
      stA_f16(ar, tid, smem + nb * ATILE);
      stB_f16(br, tid, smem + 2 * ATILE + nb * BTILE);
    }
    __syncthreads();
    buf = nb;
  }
  stage_acc(smem, wm, wn, lane, acc);
  __syncthreads();
  // Transposed write: thread -> (dn,b) pair + 32-wide s segment.
  int p = tid >> 1, dn = p & 63, b = p >> 6;
  int half = tid & 1, h = blockIdx.x;
  v8h pk[4];
#pragma unroll
  for (int c = 0; c < 4; ++c)
#pragma unroll
    for (int i = 0; i < 8; ++i)
      pk[c][i] = smem[((half * 32 + c * 8 + i) * 2 + b) * TSTR + dn];
  v8h* d = (v8h*)(out + ((size_t)(b * NHEAD + h) * HDIM + dn) * SEQ
                      + (size_t)blockIdx.y * 64 + half * 32);
  d[0] = pk[0]; d[1] = pk[1]; d[2] = pk[2]; d[3] = pk[3];
}

// ---------------------------------------------------------------------------
// Kernel 2: scores = scale * Q K^T, masked -> attn (f32). Single K-tile (dk=64).
// ---------------------------------------------------------------------------
__global__ void __launch_bounds__(256) scores_kernel(const _Float16* __restrict__ Qh,
                                                     const _Float16* __restrict__ Kh,
                                                     const int* __restrict__ mask,
                                                     const int* __restrict__ kpm,
                                                     float* __restrict__ attn) {
  __shared__ __align__(16) _Float16 smem[ATILE + BTILE];
  _Float16* sA = smem;
  _Float16* sB = smem + ATILE;
  int tid = threadIdx.x, lane = tid & 31, wave = tid >> 5;
  int wm = (wave >> 1) << 5, wn = (wave & 1) << 5;
  int bh = blockIdx.z, b = bh >> 4;
  ldA_f16_async(Qh + ((size_t)bh * SEQ + (size_t)blockIdx.y * BM) * HDIM, HDIM, tid, sA);
  ldB_f16_async(Kh + ((size_t)bh * SEQ + (size_t)blockIdx.x * BN) * HDIM, HDIM, tid, sB);
  wait_async0();
  __syncthreads();
  v8f acc[2][2] = {};
  mma_step(sA, sB, wm, wn, lane, acc);

  const float scale = 0.125f;               // 1/sqrt(64)
  int n0 = lane & 15, mh = (lane >> 4) << 3;
#pragma unroll
  for (int j = 0; j < 2; ++j) {
    int gk = blockIdx.x * BN + wn + j * 16 + n0;
    bool kz = (kpm[b * SEQ + gk] == 0);
#pragma unroll
    for (int i = 0; i < 2; ++i) {
      int gq0 = blockIdx.y * BM + wm + i * 16 + mh;
      const int* mrow = mask + (size_t)gq0 * SEQ + gk;
      float* orow = attn + ((size_t)bh * SEQ + gq0) * SEQ + gk;
#pragma unroll
      for (int r = 0; r < 8; ++r) {
        float v = acc[i][j][r] * scale;
        if (kz || mrow[(size_t)r * SEQ] == 0) v = NEGBIG;
        orow[(size_t)r * SEQ] = v;
      }
    }
  }
}

// ---------------------------------------------------------------------------
// Kernel 3: row softmax over S=2048 (one 256-thread block per row, wave32).
// ---------------------------------------------------------------------------
__global__ void __launch_bounds__(256) softmax_kernel(float* __restrict__ attn) {
  float* base = attn + (size_t)blockIdx.x * SEQ;
  int tid = threadIdx.x, lane = tid & 31, wave = tid >> 5;
  __shared__ float red[8];
  float vals[8];
  float m = -3.4e38f;
#pragma unroll
  for (int i = 0; i < 8; ++i) {
    vals[i] = base[tid + (i << 8)];
    m = fmaxf(m, vals[i]);
  }
#pragma unroll
  for (int off = 16; off > 0; off >>= 1) m = fmaxf(m, __shfl_xor(m, off, 32));
  if (lane == 0) red[wave] = m;
  __syncthreads();
  float rmax = red[0];
#pragma unroll
  for (int w = 1; w < 8; ++w) rmax = fmaxf(rmax, red[w]);
  float s = 0.0f;
#pragma unroll
  for (int i = 0; i < 8; ++i) { vals[i] = expf(vals[i] - rmax); s += vals[i]; }
#pragma unroll
  for (int off = 16; off > 0; off >>= 1) s += __shfl_xor(s, off, 32);
  __syncthreads();
  if (lane == 0) red[wave] = s;
  __syncthreads();
  float rsum = 0.0f;
#pragma unroll
  for (int w = 0; w < 8; ++w) rsum += red[w];
  float inv = 1.0f / rsum;
#pragma unroll
  for (int i = 0; i < 8; ++i) base[tid + (i << 8)] = vals[i] * inv;
}

// ---------------------------------------------------------------------------
// Kernel 4: ctx = attn(f32->f16) * V^T, double-buffered (async B, reg A).
// ---------------------------------------------------------------------------
__global__ void __launch_bounds__(256) pv_kernel(const float* __restrict__ attn,
                                                 const _Float16* __restrict__ Vt,
                                                 _Float16* __restrict__ ctx) {
  __shared__ __align__(16) _Float16 smem[2 * ATILE + 2 * BTILE];
  int tid = threadIdx.x, lane = tid & 31, wave = tid >> 5;
  int wm = (wave >> 1) << 5, wn = (wave & 1) << 5;
  int bh = blockIdx.z, b = bh >> 4, h = bh & 15;
  const float*    Ab = attn + (size_t)bh * SEQ * SEQ + (size_t)blockIdx.y * BM * SEQ;
  const _Float16* Bb = Vt + (size_t)bh * HDIM * SEQ;     // N = HDIM = one block col
  {
    ldB_f16_async(Bb, SEQ, tid, smem + 2 * ATILE);
    ARegs ar = ldA_f32_regs(Ab, SEQ, tid);
    stA_f16(ar, tid, smem);
  }
  wait_async0();
  __syncthreads();
  v8f acc[2][2] = {};
  int buf = 0;
  for (int k0 = 0; k0 < SEQ; k0 += BK) {
    int nb = buf ^ 1;
    bool more = (k0 + BK) < SEQ;
    ARegs ar;
    if (more) {
      ldB_f16_async(Bb + k0 + BK, SEQ, tid, smem + 2 * ATILE + nb * BTILE);
      ar = ldA_f32_regs(Ab + k0 + BK, SEQ, tid);
    }
    mma_step(smem + buf * ATILE, smem + 2 * ATILE + buf * BTILE, wm, wn, lane, acc);
    if (more) stA_f16(ar, tid, smem + nb * ATILE);
    wait_async0();
    __syncthreads();
    buf = nb;
  }
  stage_acc(smem, wm, wn, lane, acc);
  __syncthreads();
  int m = tid >> 1, half = tid & 1;
  int s = blockIdx.y * BM + m;              // query position
  store_row32(ctx + ((size_t)s * BATCH + b) * DMODEL + h * HDIM + half * 32,
              smem + m * TSTR + half * 32);
}

// ---------------------------------------------------------------------------
// Kernel 5: out = ctx * Wo^T + bo (f32), double-buffered (async A, reg B).
// ---------------------------------------------------------------------------
__global__ void __launch_bounds__(256) final_kernel(const _Float16* __restrict__ ctx,
                                                    const float* __restrict__ Wo,
                                                    const float* __restrict__ bo,
                                                    float* __restrict__ out) {
  __shared__ __align__(16) _Float16 smem[2 * ATILE + 2 * BTILE];
  int tid = threadIdx.x, lane = tid & 31, wave = tid >> 5;
  int wm = (wave >> 1) << 5, wn = (wave & 1) << 5;
  const _Float16* Ab = ctx + (size_t)blockIdx.y * BM * DMODEL;
  const float*    Bb = Wo + (size_t)blockIdx.x * BN * DMODEL;
  {
    ldA_f16_async(Ab, DMODEL, tid, smem);
    BRegs br = ldB_f32_regs(Bb, DMODEL, tid);
    stB_f16(br, tid, smem + 2 * ATILE);
  }
  wait_async0();
  __syncthreads();
  v8f acc[2][2] = {};
  int buf = 0;
  for (int k0 = 0; k0 < DMODEL; k0 += BK) {
    int nb = buf ^ 1;
    bool more = (k0 + BK) < DMODEL;
    BRegs br;
    if (more) {
      ldA_f16_async(Ab + k0 + BK, DMODEL, tid, smem + nb * ATILE);
      br = ldB_f32_regs(Bb + k0 + BK, DMODEL, tid);
    }
    mma_step(smem + buf * ATILE, smem + 2 * ATILE + buf * BTILE, wm, wn, lane, acc);
    if (more) stB_f16(br, tid, smem + 2 * ATILE + nb * BTILE);
    wait_async0();
    __syncthreads();
    buf = nb;
  }
  int n0 = lane & 15, mh = (lane >> 4) << 3;
#pragma unroll
  for (int j = 0; j < 2; ++j) {
    int gn = blockIdx.x * BN + wn + j * 16 + n0;
    float bj = bo[gn];
#pragma unroll
    for (int i = 0; i < 2; ++i) {
      int gm0 = blockIdx.y * BM + wm + i * 16 + mh;
      float* orow = out + (size_t)gm0 * DMODEL + gn;
#pragma unroll
      for (int r = 0; r < 8; ++r) orow[(size_t)r * DMODEL] = acc[i][j][r] + bj;
    }
  }
}

// ---------------------------------------------------------------------------
extern "C" void kernel_launch(void* const* d_in, const int* in_sizes, int n_in,
                              void* d_out, int out_size, void* d_ws, size_t ws_size,
                              hipStream_t stream) {
  const float* query = (const float*)d_in[0];
  const float* key   = (const float*)d_in[1];
  const float* value = (const float*)d_in[2];
  const int*   mask  = (const int*)d_in[3];
  const int*   kpm   = (const int*)d_in[4];
  const float* Wq    = (const float*)d_in[5];
  const float* Wk    = (const float*)d_in[6];
  const float* Wv    = (const float*)d_in[7];
  const float* Wo    = (const float*)d_in[8];
  const float* bo    = (const float*)d_in[9];

  float* out  = (float*)d_out;
  float* attn = out + (size_t)SEQ * BATCH * DMODEL;     // second tuple output

  const size_t HEADELEMS = (size_t)BATCH * NHEAD * SEQ * HDIM;  // 4M halves
  _Float16* Qh  = (_Float16*)d_ws;
  _Float16* Kh  = Qh + HEADELEMS;
  _Float16* Vt  = Kh + HEADELEMS;
  _Float16* ctx = Vt + HEADELEMS;

  dim3 blk(256);
  dim3 gproj(DMODEL / BN, (SEQ * BATCH) / BM);
  proj_qk_kernel<<<gproj, blk, 0, stream>>>(query, Wq, Qh);
  proj_qk_kernel<<<gproj, blk, 0, stream>>>(key,   Wk, Kh);
  proj_v_kernel <<<gproj, blk, 0, stream>>>(value, Wv, Vt);
  scores_kernel<<<dim3(SEQ / BN, SEQ / BM, BATCH * NHEAD), blk, 0, stream>>>(Qh, Kh, mask, kpm, attn);
  softmax_kernel<<<dim3(BATCH * NHEAD * SEQ), blk, 0, stream>>>(attn);
  pv_kernel<<<dim3(1, SEQ / BM, BATCH * NHEAD), blk, 0, stream>>>(attn, Vt, ctx);
  final_kernel<<<gproj, blk, 0, stream>>>(ctx, Wo, bo, out);
}
